// ConcatNonLinear_23837068493281
// MI455X (gfx1250) — matrix operand order repaired and checked by reading, not compiled, not run on hardware
//
#include <hip/hip_runtime.h>

typedef __attribute__((ext_vector_type(16))) __bf16 v16bf;
typedef __attribute__((ext_vector_type(8)))  float  v8f;

#define N_U_  4000
#define NN_   7000
#define NP_   7008        // NN padded to multiple of 32
#define KD_   64
#define FD_   256
#define BSZ_  4096
#define WPR_  219         // adjacency bitmask words per row (7008/32)
#define PSTR_ ((size_t)NP_ * KD_)   // partial-C stride (per K-chunk)

// ---------------------------------------------------------------------------
// Split-bf16 WMMA GEMM:  C[M,N] = act( A[M,K] @ Bt[N,K]^T + bias )
//   A : row-major fp32, lda; rows clamped to Mval-1 (garbage rows are only
//       accumulated into C rows that the store guard drops)
//   Bt: row-major fp32 [N, ldb], rows 0..N-1 valid, cols zero-padded to >= Kp
//   K-space is optionally split into nChunks chunks of chunkK (mult. of 32);
//   chunk c writes C + c*chunkStride (partials reduced by k_reduce_layer).
//   Fast region: k < (K & ~31) -> unconditional b128 loads; single guarded
//   tail iteration covers [K&~31, Kp) (Bt is zero there, A masked per elem).
// ---------------------------------------------------------------------------
__global__ __launch_bounds__(256) void k_gemm(
    const float* __restrict__ A, int lda, int Mval,
    const float* __restrict__ Bt, int ldb,
    int K, int Kp, int chunkK, int nChunks,
    float* __restrict__ C, int ldc, int chunkStride,
    int nTileM, int nTileN,
    const float* __restrict__ bias, int act)
{
  const int lane = threadIdx.x;
  const int tilesPer = nTileM * nTileN;
  const int t = blockIdx.x * blockDim.y + threadIdx.y;     // wave-uniform
  if (t >= tilesPer * nChunks) return;
  const int chunk = t / tilesPer;
  const int tile  = t % tilesPer;
  const int tm = tile / nTileN, tn = tile % nTileN;
  const int m0 = tm << 4, n0 = tn << 4;
  const int half = lane >> 4, l15 = lane & 15;

  const int kBegin = chunk * chunkK;
  const int kEnd   = min(kBegin + chunkK, Kp);
  const int kFast  = min(kEnd, K & ~31);

  const int rowA  = m0 + l15;
  const int rowAc = (rowA < Mval) ? rowA : (Mval - 1);     // clamp: branch-free
  const float* pA = A + (size_t)rowAc * lda;
  const float* pB = Bt + (size_t)(n0 + l15) * ldb;

  v8f acc = {0.f,0.f,0.f,0.f,0.f,0.f,0.f,0.f};
  int k0 = kBegin;

#pragma unroll 2
  for (; k0 < kFast; k0 += 32) {
    // A fragment (16-bit A 16x32): lane<16 -> K {0..7,16..23}; lane>=16 -> {8..15,24..31}
    const float4* qa0 = (const float4*)(pA + k0 + half * 8);
    const float4* qa1 = (const float4*)(pA + k0 + 16 + half * 8);
    float4 a0 = qa0[0], a1 = qa0[1], a2 = qa1[0], a3 = qa1[1];
    // B fragment (16-bit B 32x16): lane<16 -> K {0..7,16..23}; lane>=16 -> {8..15,24..31}
    const float4* qb0 = (const float4*)(pB + k0 + half * 8);
    const float4* qb1 = (const float4*)(pB + k0 + 16 + half * 8);
    float4 b0 = qb0[0], b1 = qb0[1], b2 = qb1[0], b3 = qb1[1];

    float af[16]  = {a0.x,a0.y,a0.z,a0.w, a1.x,a1.y,a1.z,a1.w,
                     a2.x,a2.y,a2.z,a2.w, a3.x,a3.y,a3.z,a3.w};
    float bfv[16] = {b0.x,b0.y,b0.z,b0.w, b1.x,b1.y,b1.z,b1.w,
                     b2.x,b2.y,b2.z,b2.w, b3.x,b3.y,b3.z,b3.w};

    v16bf ah, al, bh, bl;
#pragma unroll
    for (int i = 0; i < 16; ++i) {
      __bf16 h = (__bf16)af[i];  ah[i] = h;  al[i] = (__bf16)(af[i]  - (float)h);
      __bf16 g = (__bf16)bfv[i]; bh[i] = g;  bl[i] = (__bf16)(bfv[i] - (float)g);
    }
    acc = __builtin_amdgcn_wmma_f32_16x16x32_bf16(false, ah, false, bh, (short)0, acc, false, false);
    acc = __builtin_amdgcn_wmma_f32_16x16x32_bf16(false, ah, false, bl, (short)0, acc, false, false);
    acc = __builtin_amdgcn_wmma_f32_16x16x32_bf16(false, al, false, bh, (short)0, acc, false, false);
  }

  // guarded tail: covers [K & ~31, Kp); Bt rows are zero-padded there
  for (; k0 < kEnd; k0 += 32) {
    float af[16], bfv[16];
    const int ka0 = k0 + half * 8, ka1 = k0 + 16 + half * 8;
#pragma unroll
    for (int i = 0; i < 8; ++i) {
      af[i]     = (ka0 + i < K) ? pA[ka0 + i] : 0.0f;
      af[8 + i] = (ka1 + i < K) ? pA[ka1 + i] : 0.0f;
    }
    const float4* qb0 = (const float4*)(pB + k0 + half * 8);
    const float4* qb1 = (const float4*)(pB + k0 + 16 + half * 8);
    float4 b0 = qb0[0], b1 = qb0[1], b2 = qb1[0], b3 = qb1[1];
    bfv[0]=b0.x;  bfv[1]=b0.y;  bfv[2]=b0.z;  bfv[3]=b0.w;
    bfv[4]=b1.x;  bfv[5]=b1.y;  bfv[6]=b1.z;  bfv[7]=b1.w;
    bfv[8]=b2.x;  bfv[9]=b2.y;  bfv[10]=b2.z; bfv[11]=b2.w;
    bfv[12]=b3.x; bfv[13]=b3.y; bfv[14]=b3.z; bfv[15]=b3.w;

    v16bf ah, al, bh, bl;
#pragma unroll
    for (int i = 0; i < 16; ++i) {
      __bf16 h = (__bf16)af[i];  ah[i] = h;  al[i] = (__bf16)(af[i]  - (float)h);
      __bf16 g = (__bf16)bfv[i]; bh[i] = g;  bl[i] = (__bf16)(bfv[i] - (float)g);
    }
    acc = __builtin_amdgcn_wmma_f32_16x16x32_bf16(false, ah, false, bh, (short)0, acc, false, false);
    acc = __builtin_amdgcn_wmma_f32_16x16x32_bf16(false, ah, false, bl, (short)0, acc, false, false);
    acc = __builtin_amdgcn_wmma_f32_16x16x32_bf16(false, al, false, bh, (short)0, acc, false, false);
  }

  float* Cc = C + (size_t)chunk * chunkStride;
  const int col = n0 + l15;
  const float bv = bias ? bias[col] : 0.0f;
#pragma unroll
  for (int v = 0; v < 8; ++v) {
    int row = m0 + v + half * 8;            // C/D layout: VGPR v -> row m0+v (+8 upper half)
    float val = acc[v] + bv;
    if (act == 1) val = 0.5f * val * (1.0f + erff(val * 0.70710678118654752440f));
    if (row < Mval) Cc[(size_t)row * ldc + col] = val;
  }
}

// ---------------------------------------------------------------------------
// E0t[k, i] = concat(E_u,E_v)[i, k] (zero-padded i>=NN), also h[:,0:64] = E0
// ---------------------------------------------------------------------------
__global__ void k_build_e0t(const float* __restrict__ Eu, const float* __restrict__ Ev,
                            float* __restrict__ E0t, float* __restrict__ h)
{
  int idx = blockIdx.x * blockDim.x + threadIdx.x;
  if (idx >= KD_ * NP_) return;
  int i = idx % NP_, k = idx / NP_;
  float v = 0.0f;
  if (i < N_U_)      v = Eu[(size_t)i * KD_ + k];
  else if (i < NN_)  v = Ev[(size_t)(i - N_U_) * KD_ + k];
  E0t[(size_t)k * NP_ + i] = v;
  if (i < NN_) h[(size_t)i * FD_ + k] = v;
}

// Reduce the 4 K-chunk partials -> h[:, off:off+64]; optionally also emit the
// transposed (zero-padded) Et for the next layer. Fixed order => deterministic.
__global__ void k_reduce_layer(const float* __restrict__ part, float* __restrict__ h,
                               int off, float* __restrict__ Et, int writeEt)
{
  int idx = blockIdx.x * blockDim.x + threadIdx.x;
  if (idx >= KD_ * NP_) return;
  int i = idx % NP_, k = idx / NP_;
  if (i >= NN_) {
    if (writeEt) Et[(size_t)k * NP_ + i] = 0.0f;
    return;
  }
  size_t o = (size_t)i * KD_ + k;
  float s = ((part[o] + part[PSTR_ + o]) + part[2 * PSTR_ + o]) + part[3 * PSTR_ + o];
  h[(size_t)i * FD_ + off + k] = s;
  if (writeEt) Et[(size_t)k * NP_ + i] = s;
}

// out[C,R] = in[R,C]^T
__global__ void k_transpose(const float* __restrict__ in, float* __restrict__ out, int R, int C)
{
  int idx = blockIdx.x * blockDim.x + threadIdx.x;
  if (idx >= R * C) return;
  int r = idx % R, c = idx / R;
  out[(size_t)c * R + r] = in[(size_t)r * C + c];
}

// s1[i] = Wh[i,:]·a1 + ba1 ; s2[i] = Wh[i,:]·a2 + ba2  (one wave per row)
__global__ __launch_bounds__(256) void k_s1s2(
    const float* __restrict__ Wh,
    const float* __restrict__ a1, const float* __restrict__ ba1,
    const float* __restrict__ a2, const float* __restrict__ ba2,
    float* __restrict__ s1, float* __restrict__ s2)
{
  int row = blockIdx.x * blockDim.y + threadIdx.y;
  if (row >= NN_) return;
  int lane = threadIdx.x;
  float acc1 = 0.f, acc2 = 0.f;
  const float* p = Wh + (size_t)row * FD_;
#pragma unroll
  for (int c = 0; c < FD_ / 32; ++c) {
    float w = p[lane + 32 * c];
    acc1 += w * a1[lane + 32 * c];
    acc2 += w * a2[lane + 32 * c];
  }
#pragma unroll
  for (int off = 16; off > 0; off >>= 1) {
    acc1 += __shfl_xor(acc1, off);
    acc2 += __shfl_xor(acc2, off);
  }
  if (lane == 0) { s1[row] = acc1 + ba1[0]; s2[row] = acc2 + ba2[0]; }
}

__global__ void k_zero_u32(unsigned* __restrict__ p, int n)
{
  int idx = blockIdx.x * blockDim.x + threadIdx.x;
  if (idx < n) p[idx] = 0u;
}

// adjacency bitmask scatter (idempotent -> deterministic)
__global__ void k_scatter(const int* __restrict__ users, const int* __restrict__ items,
                          unsigned* __restrict__ bits)
{
  int b = blockIdx.x * blockDim.x + threadIdx.x;
  if (b >= BSZ_) return;
  int u = users[b];
  int v = items[b] + N_U_;
  atomicOr(&bits[(size_t)u * WPR_ + (v >> 5)], 1u << (v & 31));
  atomicOr(&bits[(size_t)v * WPR_ + (u >> 5)], 1u << (u & 31));
}

// deterministic column sums of Wh (two stage, no float atomics)
__global__ __launch_bounds__(256) void k_colsum_part(const float* __restrict__ Wh,
                                                     float* __restrict__ part)
{
  int t = threadIdx.x;
  float acc = 0.f;
  for (int i = blockIdx.x; i < NN_; i += 64) acc += Wh[(size_t)i * FD_ + t];
  part[(size_t)blockIdx.x * FD_ + t] = acc;
}
__global__ __launch_bounds__(256) void k_colsum_fin(const float* __restrict__ part,
                                                    float* __restrict__ csum)
{
  int t = threadIdx.x;
  float s = 0.f;
#pragma unroll
  for (int b = 0; b < 64; ++b) s += part[(size_t)b * FD_ + t];
  csum[t] = s;
}

// Sparse GAT row: exact softmax over edges (non-edges underflow to 0 in ref),
// edge-less rows -> uniform 1/NN -> H[i] = colmean(Wh). One wave per row.
__global__ __launch_bounds__(256) void k_attn(
    const unsigned* __restrict__ bits,
    const float* __restrict__ s1, const float* __restrict__ s2,
    const float* __restrict__ Wh, const float* __restrict__ csum,
    float* __restrict__ H)
{
  int row = blockIdx.x * blockDim.y + threadIdx.y;
  if (row >= NN_) return;
  int lane = threadIdx.x;
  const unsigned* rb = bits + (size_t)row * WPR_;
  const float si = s1[row];

  // pass 1: edge count + max (redundant across lanes, wave-uniform)
  float m = -3.0e38f;
  int cnt = 0;
  for (int w = 0; w < WPR_; ++w) {
    unsigned bw = rb[w];
    while (bw) {
      int b = __builtin_ctz(bw); bw &= bw - 1;
      int j = (w << 5) + b;
      float v = si + s2[j];
      v = (v >= 0.f) ? v : 0.2f * v;          // leaky_relu, ALPHA=0.2
      m = fmaxf(m, v);
      ++cnt;
    }
  }

  if (cnt == 0) {
    const float inv = 1.0f / 7000.0f;
#pragma unroll
    for (int c = 0; c < FD_ / 32; ++c)
      H[(size_t)row * FD_ + lane + 32 * c] = csum[lane + 32 * c] * inv;
    return;
  }

  // pass 2: softmax denominator
  float denom = 0.f;
  for (int w = 0; w < WPR_; ++w) {
    unsigned bw = rb[w];
    while (bw) {
      int b = __builtin_ctz(bw); bw &= bw - 1;
      int j = (w << 5) + b;
      float v = si + s2[j];
      v = (v >= 0.f) ? v : 0.2f * v;
      denom += expf(v - m);
    }
  }

  // pass 3: weighted aggregation of Wh rows
  float acc[FD_ / 32];
#pragma unroll
  for (int c = 0; c < FD_ / 32; ++c) acc[c] = 0.f;
  for (int w = 0; w < WPR_; ++w) {
    unsigned bw = rb[w];
    while (bw) {
      int b = __builtin_ctz(bw); bw &= bw - 1;
      int j = (w << 5) + b;
      float v = si + s2[j];
      v = (v >= 0.f) ? v : 0.2f * v;
      float wq = expf(v - m) / denom;
      const float* pw = Wh + (size_t)j * FD_;
#pragma unroll
      for (int c = 0; c < FD_ / 32; ++c) acc[c] += wq * pw[lane + 32 * c];
    }
  }
#pragma unroll
  for (int c = 0; c < FD_ / 32; ++c)
    H[(size_t)row * FD_ + lane + 32 * c] = acc[c];
}

// x[b] = [ H[users[b]] , H[items[b]+N_U] ]
__global__ void k_gather(const int* __restrict__ users, const int* __restrict__ items,
                         const float* __restrict__ H, float* __restrict__ x)
{
  int idx = blockIdx.x * blockDim.x + threadIdx.x;
  if (idx >= BSZ_ * FD_) return;
  int b = idx / FD_, c = idx % FD_;
  x[(size_t)b * (2 * FD_) + c]       = H[(size_t)users[b] * FD_ + c];
  x[(size_t)b * (2 * FD_) + FD_ + c] = H[(size_t)(items[b] + N_U_) * FD_ + c];
}

// out[b] = y2[b,:]·W3 + b3
__global__ void k_out(const float* __restrict__ y2, const float* __restrict__ W3,
                      const float* __restrict__ b3, float* __restrict__ out)
{
  int b = blockIdx.x * blockDim.x + threadIdx.x;
  if (b >= BSZ_) return;
  const float* p = y2 + (size_t)b * KD_;
  float s = 0.f;
#pragma unroll
  for (int k = 0; k < KD_; ++k) s += p[k] * W3[k];
  out[b] = s + b3[0];
}

// ---------------------------------------------------------------------------
extern "C" void kernel_launch(void* const* d_in, const int* in_sizes, int n_in,
                              void* d_out, int out_size, void* d_ws, size_t ws_size,
                              hipStream_t stream)
{
  const int*   users = (const int*)  d_in[0];
  const int*   items = (const int*)  d_in[1];
  const float* A     = (const float*)d_in[2];   // [7000,7000]
  const float* Eu    = (const float*)d_in[3];
  const float* Ev    = (const float*)d_in[4];
  const float* W     = (const float*)d_in[5];
  const float* bW    = (const float*)d_in[6];
  const float* a1    = (const float*)d_in[7];
  const float* ba1   = (const float*)d_in[8];
  const float* a2    = (const float*)d_in[9];
  const float* ba2   = (const float*)d_in[10];
  const float* W1    = (const float*)d_in[11];  // [512,128]
  const float* b1    = (const float*)d_in[12];
  const float* W2    = (const float*)d_in[13];  // [128,64]
  const float* b2    = (const float*)d_in[14];
  const float* W3    = (const float*)d_in[15];  // [64]
  const float* b3    = (const float*)d_in[16];
  float* out = (float*)d_out;

  // workspace carve-out (all offsets 64B aligned)
  float* wsp = (float*)d_ws;
  auto alloc = [&](size_t n) { float* p = wsp; wsp += (n + 15) & ~(size_t)15; return p; };
  float*    Et   = alloc((size_t)KD_ * NP_);        // transposed E_l, zero-padded
  float*    h    = alloc((size_t)NN_ * FD_);        // [E0|E1|E2|E3]
  float*    p4   = alloc(4 * PSTR_);                // K-chunk partials
  float*    Wt   = alloc((size_t)FD_ * FD_);
  float*    Wh   = alloc((size_t)NN_ * FD_);
  float*    s1   = alloc(NN_);
  float*    s2   = alloc(NN_);
  float*    part = alloc((size_t)64 * FD_);
  float*    csum = alloc(FD_);
  unsigned* bits = (unsigned*)alloc((size_t)NN_ * WPR_);
  float*    H    = alloc((size_t)NN_ * FD_);
  float*    x    = alloc((size_t)BSZ_ * 2 * FD_);
  float*    W1t  = alloc((size_t)128 * 512);
  float*    y1   = alloc((size_t)BSZ_ * 128);
  float*    W2t  = alloc((size_t)64 * 128);
  float*    y2   = alloc((size_t)BSZ_ * KD_);

  const dim3 wblk(32, 8);   // 8 waves / block, one 16x16 C-tile per wave
  const int TKN = KD_ * NP_;

  // ---- LightGCN: E_{l+1} = A @ E_l, h = concat(layers) ----
  k_build_e0t<<<(TKN + 255) / 256, 256, 0, stream>>>(Eu, Ev, Et, h);
  for (int l = 0; l < 3; ++l) {
    // K split into 4 chunks of 1760 (mult. of 32; last = 1728) -> 7008 waves
    k_gemm<<<(438 * 4 * 4) / 8, wblk, 0, stream>>>(
        A, NN_, NN_, Et, NP_, NN_, NP_, /*chunkK=*/1760, /*nChunks=*/4,
        p4, KD_, (int)PSTR_, 438, 4, nullptr, 0);
    k_reduce_layer<<<(TKN + 255) / 256, 256, 0, stream>>>(
        p4, h, KD_ * (l + 1), Et, (l < 2) ? 1 : 0);
  }

  // ---- GAT: Wh = h @ W + bW ; s1,s2 ----
  k_transpose<<<(FD_ * FD_ + 255) / 256, 256, 0, stream>>>(W, Wt, FD_, FD_);
  k_gemm<<<(438 * 16) / 8, wblk, 0, stream>>>(
      h, FD_, NN_, Wt, FD_, FD_, FD_, /*chunkK=*/FD_, /*nChunks=*/1,
      Wh, FD_, 0, 438, 16, bW, 0);
  k_s1s2<<<(NN_ + 7) / 8, wblk, 0, stream>>>(Wh, a1, ba1, a2, ba2, s1, s2);

  // ---- sparse attention ----
  const int nbits = NN_ * WPR_;
  k_zero_u32<<<(nbits + 255) / 256, 256, 0, stream>>>(bits, nbits);
  k_scatter<<<(BSZ_ + 255) / 256, 256, 0, stream>>>(users, items, bits);
  k_colsum_part<<<64, 256, 0, stream>>>(Wh, part);
  k_colsum_fin<<<1, 256, 0, stream>>>(part, csum);
  k_attn<<<(NN_ + 7) / 8, wblk, 0, stream>>>(bits, s1, s2, Wh, csum, H);

  // ---- MLP head ----
  k_gather<<<(BSZ_ * FD_ + 255) / 256, 256, 0, stream>>>(users, items, H, x);
  k_transpose<<<(512 * 128 + 255) / 256, 256, 0, stream>>>(W1, W1t, 512, 128);
  k_gemm<<<(256 * 8) / 8, wblk, 0, stream>>>(
      x, 512, BSZ_, W1t, 512, 512, 512, /*chunkK=*/512, /*nChunks=*/1,
      y1, 128, 0, 256, 8, b1, 1);
  k_transpose<<<(128 * 64 + 255) / 256, 256, 0, stream>>>(W2, W2t, 128, 64);
  k_gemm<<<(256 * 4) / 8, wblk, 0, stream>>>(
      y1, 128, BSZ_, W2t, 128, 128, 128, /*chunkK=*/128, /*nChunks=*/1,
      y2, KD_, 0, 256, 4, b2, 1);
  k_out<<<(BSZ_ + 255) / 256, 256, 0, stream>>>(y2, W3, b3, out);
}
